// SenriAttention_53979148976313
// MI455X (gfx1250) — compile-verified, matrix operand order checked by reading, not run on hardware
//
#include <hip/hip_runtime.h>
#include <hip/hip_bf16.h>

// ---------------------------------------------------------------------------
// Fast-weight linear-attention step for MI455X (gfx1250, wave32, WMMA + TDM).
// Heavy math: bf16 WMMA (v_wmma_f32_16x16x32_bf16) with f32 accumulation.
// Operand staging: Tensor Data Mover (tensor_load_to_lds) double-buffered LDS.
// ---------------------------------------------------------------------------

typedef __attribute__((ext_vector_type(16))) __bf16 v16bf;
typedef __attribute__((ext_vector_type(8)))  __bf16 v8bf;
typedef __attribute__((ext_vector_type(4)))  __bf16 v4bf;
typedef __attribute__((ext_vector_type(8)))  float  v8f;
typedef __attribute__((ext_vector_type(4)))  unsigned int u32x4;
typedef __attribute__((ext_vector_type(8)))  int        i32x8;
typedef __attribute__((ext_vector_type(4)))  int        i32x4;

#define D_DIM 2048
#define BS_DIM 8192   // B*S = 2*4096
#define TILE 128
#define PANEL 64      // K-panel staged per TDM transfer (2 WMMA K-steps)

#if __has_builtin(__builtin_amdgcn_tensor_load_to_lds)
#define HAVE_TDM 1
#else
#define HAVE_TDM 0
#endif

__device__ __forceinline__ float phi_act(float x) {
    // elu(x)+1 == x+1 (x>0) else exp(x)
    return x > 0.0f ? x + 1.0f : __expf(x);
}

// Load a 16x32 (rows x K) bf16 fragment from a K-major (row-major [rows,K])
// matrix (global or LDS), matching the documented 16-bit A/B operand layout:
//   lanes 0-15 : row=row0+lane,    halves 0-7 = K[k0..k0+7],    8-15 = K[k0+16..+23]
//   lanes 16-31: row=row0+lane-16, halves 0-7 = K[k0+8..k0+15], 8-15 = K[k0+24..+31]
__device__ __forceinline__ v16bf load_frag(const __bf16* __restrict__ P,
                                           int row0, int k0, int ld, int lane) {
    int r  = row0 + (lane & 15);
    int kh = (lane >> 4) << 3;            // 0 or 8
    const __bf16* base = P + (size_t)r * ld + k0 + kh;
    v8bf lo = *(const v8bf*)(base);       // 16B aligned (k0%32==0, ld%8==0)
    v8bf hi = *(const v8bf*)(base + 16);
    v16bf f;
#pragma unroll
    for (int t = 0; t < 8; ++t) { f[t] = lo[t]; f[t + 8] = hi[t]; }
    return f;
}

#if HAVE_TDM
// Issue one TDM 2D tile transfer: [tileRows x tileK] bf16 tile of a
// [rowsTotal x Kelems] row-major tensor (row pitch = strideElems) -> LDS.
__device__ __forceinline__ void tdm_load_2d(unsigned lds_off, const __bf16* gsrc,
                                            unsigned Kelems, unsigned rowsTotal,
                                            unsigned tileK, unsigned tileRows,
                                            unsigned long long strideElems) {
    unsigned long long ga = (unsigned long long)(uintptr_t)gsrc;
    u32x4 g0;
    g0[0] = 1u;                                            // count=1 valid descr
    g0[1] = lds_off;                                       // lds_addr
    g0[2] = (unsigned)(ga & 0xFFFFFFFFu);                  // global_addr[31:0]
    g0[3] = (unsigned)((ga >> 32) & 0x01FFFFFFu) | (2u << 30); // [56:32], type=2
    i32x8 g1;
    g1[0] = (int)(1u << 16);                               // wg_mask=0, data_size=2B
    g1[1] = (int)((Kelems & 0xFFFFu) << 16);               // tensor_dim0 lo
    g1[2] = (int)((Kelems >> 16) | ((rowsTotal & 0xFFFFu) << 16)); // dim0 hi|dim1 lo
    g1[3] = (int)((rowsTotal >> 16) | (tileK << 16));      // dim1 hi | tile_dim0
    g1[4] = (int)tileRows;                                 // tile_dim1, tile_dim2=0
    g1[5] = (int)(strideElems & 0xFFFFFFFFull);            // dim0_stride lo
    g1[6] = (int)((strideElems >> 32) & 0xFFFFull);        // dim0_stride hi
    g1[7] = 0;
    i32x4 z4 = {0, 0, 0, 0};
#if __clang_major__ >= 23
    i32x8 z8 = {0, 0, 0, 0, 0, 0, 0, 0};
    __builtin_amdgcn_tensor_load_to_lds(g0, g1, z4, z4, z8, 0);
#else
    __builtin_amdgcn_tensor_load_to_lds(g0, g1, z4, z4, 0);
#endif
}
#endif

// ---------------------------------------------------------------------------
// Conversions
// ---------------------------------------------------------------------------
__global__ void cvt4_f32_bf16(const float* __restrict__ in,
                              __bf16* __restrict__ out, int n4) {
    int i = blockIdx.x * blockDim.x + threadIdx.x;
    if (i < n4) {
        float4 f = ((const float4*)in)[i];
        v4bf o;
        o[0] = (__bf16)f.x; o[1] = (__bf16)f.y;
        o[2] = (__bf16)f.z; o[3] = (__bf16)f.w;
        ((v4bf*)out)[i] = o;
    }
}

// out[e*dim + d] = (bf16) in[d*dim + e]   (LDS-tiled transpose)
__global__ void cvt_tr_f32_bf16(const float* __restrict__ in,
                                __bf16* __restrict__ out, int dim) {
    __shared__ float tile[32][33];
    int bx = blockIdx.x * 32, by = blockIdx.y * 32;
    int tx = threadIdx.x, ty = threadIdx.y;   // block (32,8)
#pragma unroll
    for (int j = 0; j < 32; j += 8)
        tile[ty + j][tx] = in[(size_t)(by + ty + j) * dim + bx + tx];
    __syncthreads();
#pragma unroll
    for (int j = 0; j < 32; j += 8)
        out[(size_t)(bx + ty + j) * dim + by + tx] = (__bf16)tile[tx][ty + j];
}

// ---------------------------------------------------------------------------
// Generic bf16 WMMA GEMM:  C[M,N] = A[M,K] * B[N,K]^T  (both K-major)
// 128x128 block tile, 8 waves (4x2), wave tile 32x64 = 2x4 WMMA tiles.
// K-panels of 64 staged into double-buffered LDS by the Tensor Data Mover.
// Epilogue (compile time):
//   0: bf16 phi(acc), row-major          (phi_q)
//   1: bf16 phi(acc), transposed         (phi_k^T)
//   2: bf16 acc,      transposed         (v^T)
//   3: bf16 acc/denom[row], row-major    (retrieved)
//   4: f32  acc, row-major               (output)
//   5: f32  acc + Cin, row-major         (M_new)
// ---------------------------------------------------------------------------
template <int MODE>
__global__ __launch_bounds__(256, 1)
void gemm_bf16_wmma(const __bf16* __restrict__ A, const __bf16* __restrict__ B,
                    int Mtot, int Ntot, int K,
                    __bf16* __restrict__ outb, int ldo,
                    float* __restrict__ outf,
                    const float* __restrict__ denom,
                    const float* __restrict__ Cin) {
    __shared__ __bf16 sA[2][TILE][PANEL];   // 16KB x2
    __shared__ __bf16 sB[2][TILE][PANEL];   // 16KB x2

    const int lane = threadIdx.x & 31;
    const int wave = threadIdx.x >> 5;        // 0..7
    const int wm = wave & 3, wn = wave >> 2;  // 4 x 2 wave grid
    const int mBlk = blockIdx.y * TILE;
    const int nBlk = blockIdx.x * TILE;
    const int m0 = mBlk + wm * 32;
    const int n0 = nBlk + wn * 64;

    v8f c[2][4] = {};

    const int nPanels = K / PANEL;

#if HAVE_TDM
    if (wave == 0) {
        tdm_load_2d((unsigned)(uintptr_t)&sA[0][0][0], A + (size_t)mBlk * K,
                    (unsigned)K, (unsigned)Mtot, PANEL, TILE, (unsigned long long)K);
        tdm_load_2d((unsigned)(uintptr_t)&sB[0][0][0], B + (size_t)nBlk * K,
                    (unsigned)K, (unsigned)Ntot, PANEL, TILE, (unsigned long long)K);
    }
#endif

    for (int p = 0; p < nPanels; ++p) {
#if HAVE_TDM
        const int buf = p & 1;
        __builtin_amdgcn_s_wait_tensorcnt(0);   // no-op for waves with cnt==0
        __syncthreads();                        // publish buf; prev readers done
        if (p + 1 < nPanels && wave == 0) {
            const int k1 = (p + 1) * PANEL;
            tdm_load_2d((unsigned)(uintptr_t)&sA[buf ^ 1][0][0],
                        A + (size_t)mBlk * K + k1, (unsigned)K, (unsigned)Mtot,
                        PANEL, TILE, (unsigned long long)K);
            tdm_load_2d((unsigned)(uintptr_t)&sB[buf ^ 1][0][0],
                        B + (size_t)nBlk * K + k1, (unsigned)K, (unsigned)Ntot,
                        PANEL, TILE, (unsigned long long)K);
        }
#else
        const int buf = 0;
        const int k0g = p * PANEL;
        __syncthreads();
        // cooperative copy: 128 rows x 128B per operand, 256 threads x 4 uint4
        {
            const int tid = threadIdx.x;
            for (int u = tid; u < (TILE * PANEL * 2 / 16); u += 256) {
                int row = u >> 3, c16 = u & 7;
                ((uint4*)&sA[0][row][0])[c16] =
                    ((const uint4*)(A + (size_t)(mBlk + row) * K + k0g))[c16];
                ((uint4*)&sB[0][row][0])[c16] =
                    ((const uint4*)(B + (size_t)(nBlk + row) * K + k0g))[c16];
            }
        }
        __syncthreads();
#endif
        // 2 WMMA K-steps out of the staged panel
#pragma unroll
        for (int kk = 0; kk < PANEL; kk += 32) {
            v16bf a[2], b[4];
#pragma unroll
            for (int i = 0; i < 2; ++i)
                a[i] = load_frag(&sA[buf][0][0], wm * 32 + 16 * i, kk, PANEL, lane);
#pragma unroll
            for (int j = 0; j < 4; ++j)
                b[j] = load_frag(&sB[buf][0][0], wn * 64 + 16 * j, kk, PANEL, lane);
#pragma unroll
            for (int i = 0; i < 2; ++i)
#pragma unroll
                for (int j = 0; j < 4; ++j)
                    c[i][j] = __builtin_amdgcn_wmma_f32_16x16x32_bf16(
                        false, a[i], false, b[j], (short)0, c[i][j], false, false);
        }
#if HAVE_TDM
        __syncthreads();   // readers done before this buf is refilled at p+2
#endif
    }

    // C/D layout: lanes 0-15 -> N=lane, elems r -> M=r ; lanes 16-31 -> M=r+8.
    const int half  = lane >> 4;     // 0/1
    const int col_l = lane & 15;
#pragma unroll
    for (int i = 0; i < 2; ++i) {
#pragma unroll
        for (int j = 0; j < 4; ++j) {
            const int rowbase = m0 + i * 16 + half * 8;
            const int col     = n0 + j * 16 + col_l;
            if (MODE == 1 || MODE == 2) {
                // transposed store: 8 consecutive rows per lane -> one 16B store
                v8bf pk;
#pragma unroll
                for (int r = 0; r < 8; ++r) {
                    float v = c[i][j][r];
                    if (MODE == 1) v = phi_act(v);
                    pk[r] = (__bf16)v;
                }
                *(v8bf*)(outb + (size_t)col * ldo + rowbase) = pk;
            } else {
#pragma unroll
                for (int r = 0; r < 8; ++r) {
                    const int row = rowbase + r;
                    const float v = c[i][j][r];
                    if (MODE == 0)
                        outb[(size_t)row * ldo + col] = (__bf16)phi_act(v);
                    else if (MODE == 3)
                        outb[(size_t)row * ldo + col] = (__bf16)(v / denom[row]);
                    else if (MODE == 4)
                        outf[(size_t)row * Ntot + col] = v;
                    else if (MODE == 5)
                        outf[(size_t)row * Ntot + col] =
                            v + Cin[(size_t)row * Ntot + col];
                }
            }
        }
    }
}

// ---------------------------------------------------------------------------
// denom[m] = sum_d phi_q[m,d] * z[d] + eps       (one wave per row)
// ---------------------------------------------------------------------------
__global__ void denom_kernel(const __bf16* __restrict__ phi_q,
                             const float* __restrict__ z,
                             float* __restrict__ denom, int K) {
    int row  = blockIdx.x * 8 + (threadIdx.x >> 5);
    int lane = threadIdx.x & 31;
    const __bf16* p = phi_q + (size_t)row * K;
    float s = 0.0f;
    for (int k = lane; k < K; k += 32) s += (float)p[k] * z[k];
#pragma unroll
    for (int off = 16; off > 0; off >>= 1) s += __shfl_down(s, off, 32);
    if (lane == 0) denom[row] = s + 1e-6f;
}

// ---------------------------------------------------------------------------
// z_new[d] = z[d] + sum_m phi_kT[d,m]            (one wave per row of phi_kT)
// ---------------------------------------------------------------------------
__global__ void znew_kernel(const __bf16* __restrict__ phi_kT,
                            const float* __restrict__ z,
                            float* __restrict__ zout, int BS) {
    int d    = blockIdx.x * 8 + (threadIdx.x >> 5);
    int lane = threadIdx.x & 31;
    const __bf16* p = phi_kT + (size_t)d * BS;
    float s = 0.0f;
    for (int m = lane; m < BS; m += 32) s += (float)p[m];
#pragma unroll
    for (int off = 16; off > 0; off >>= 1) s += __shfl_down(s, off, 32);
    if (lane == 0) zout[d] = z[d] + s;
}

// ---------------------------------------------------------------------------
extern "C" void kernel_launch(void* const* d_in, const int* in_sizes, int n_in,
                              void* d_out, int out_size, void* d_ws, size_t ws_size,
                              hipStream_t stream) {
    const float* x  = (const float*)d_in[0];
    const float* Wq = (const float*)d_in[1];
    const float* Wk = (const float*)d_in[2];
    const float* Wv = (const float*)d_in[3];
    const float* Wo = (const float*)d_in[4];
    const float* Mm = (const float*)d_in[5];
    const float* z  = (const float*)d_in[6];
    float* out = (float*)d_out;

    constexpr int    D  = D_DIM;
    constexpr int    BS = BS_DIM;
    constexpr size_t XE = (size_t)BS * D;   // 16,777,216 elements
    constexpr size_t WE = (size_t)D * D;    //  4,194,304 elements

    // workspace layout (bytes)
    char* p = (char*)d_ws;
    constexpr size_t MB = 1024ull * 1024ull;
    __bf16* Xb    = (__bf16*)(p + 0);            // 32 MB (aliased by retr later)
    __bf16* Wqb   = (__bf16*)(p + 32 * MB);      //  8 MB
    __bf16* Wkb   = (__bf16*)(p + 40 * MB);
    __bf16* Wvb   = (__bf16*)(p + 48 * MB);
    __bf16* Wob   = (__bf16*)(p + 56 * MB);
    __bf16* MbT   = (__bf16*)(p + 64 * MB);      //  8 MB, M transposed
    __bf16* phiQ  = (__bf16*)(p + 72 * MB);      // 32 MB
    __bf16* phiKT = (__bf16*)(p + 104 * MB);     // 32 MB  [D, BS]
    __bf16* vT    = (__bf16*)(p + 136 * MB);     // 32 MB  [D, BS]
    float*  den   = (float*)(p + 168 * MB);      // 32 KB
    __bf16* retr  = Xb;                          // X no longer needed by then

    // --- precision conversion -------------------------------------------
    cvt4_f32_bf16<<<(int)(XE / 4 / 256), 256, 0, stream>>>(x, Xb, (int)(XE / 4));
    cvt4_f32_bf16<<<(int)(WE / 4 / 256), 256, 0, stream>>>(Wq, Wqb, (int)(WE / 4));
    cvt4_f32_bf16<<<(int)(WE / 4 / 256), 256, 0, stream>>>(Wk, Wkb, (int)(WE / 4));
    cvt4_f32_bf16<<<(int)(WE / 4 / 256), 256, 0, stream>>>(Wv, Wvb, (int)(WE / 4));
    cvt4_f32_bf16<<<(int)(WE / 4 / 256), 256, 0, stream>>>(Wo, Wob, (int)(WE / 4));
    cvt_tr_f32_bf16<<<dim3(D / 32, D / 32), dim3(32, 8), 0, stream>>>(Mm, MbT, D);

    dim3 blk(256);
    dim3 gBig(D / TILE, BS / TILE);   // 16 x 64 blocks
    dim3 gSq(D / TILE, D / TILE);     // 16 x 16 blocks

    // phi_q = phi(X Wq^T)                       [BS, D] row-major
    gemm_bf16_wmma<0><<<gBig, blk, 0, stream>>>(Xb, Wqb, BS, D, D,
                                                phiQ, D, nullptr, nullptr, nullptr);
    // phi_k^T = phi(X Wk^T)^T                   [D, BS]
    gemm_bf16_wmma<1><<<gBig, blk, 0, stream>>>(Xb, Wkb, BS, D, D,
                                                phiKT, BS, nullptr, nullptr, nullptr);
    // v^T = (X Wv^T)^T                          [D, BS]
    gemm_bf16_wmma<2><<<gBig, blk, 0, stream>>>(Xb, Wvb, BS, D, D,
                                                vT, BS, nullptr, nullptr, nullptr);
    // denom[m] = phi_q . z + eps
    denom_kernel<<<BS / 8, 256, 0, stream>>>(phiQ, z, den, D);
    // retrieved = (phi_q M) / denom             [BS, D]
    gemm_bf16_wmma<3><<<gBig, blk, 0, stream>>>(phiQ, MbT, BS, D, D,
                                                retr, D, nullptr, den, nullptr);
    // output = retrieved Wo^T  -> d_out[0 .. XE)
    gemm_bf16_wmma<4><<<gBig, blk, 0, stream>>>(retr, Wob, BS, D, D,
                                                nullptr, 0, out, nullptr, nullptr);
    // M_new = M + phi_k^T v    -> d_out[XE .. XE+WE)
    gemm_bf16_wmma<5><<<gSq, blk, 0, stream>>>(phiKT, vT, D, D, BS,
                                               nullptr, 0, out + XE, nullptr, Mm);
    // z_new                    -> d_out[XE+WE .. XE+WE+D)
    znew_kernel<<<D / 8, 256, 0, stream>>>(phiKT, z, out + XE + WE, BS);
}
